// GATCommunityDetector_11261404250471
// MI455X (gfx1250) — matrix operand order repaired and checked by reading, not compiled, not run on hardware
//
#include <hip/hip_runtime.h>

#define N_NODES 50000
#define N_EDGES 800000
#define ETOT    850000   // E + N self-loops
#define DIN     256
#define HIDC    128
#define NHEAD   4
#define CHEAD   32
#define BN_EPS  1e-5f
#define GEMM_GRID 1024   // persistent blocks; grid-stride over 3125 row tiles

typedef __attribute__((ext_vector_type(16))) _Float16 v16h;
typedef __attribute__((ext_vector_type(8)))  _Float16 v8h;
typedef __attribute__((ext_vector_type(8)))  float    v8f;

// ---------------------------------------------------------------------------
// One-time W (f32, [K,128]) -> f16 fragment-ordered Wf:
//   Wf[ ((ct*K32 + kk)*32 + lane)*16 + e ] = W[(kk*32 + 16*(lane>>4) + e)*128 + ct*16 + (lane&15)]
// so a wave's B fragment for k-step kk is 32 contiguous bytes per lane.
// ---------------------------------------------------------------------------
__global__ __launch_bounds__(256)
void convert_w(const float* __restrict__ W, _Float16* __restrict__ Wf, int K32)
{
    int idx = blockIdx.x * 256 + threadIdx.x;
    int total = K32 * 32 * 16 * 8;            // == K * 128
    if (idx >= total) return;
    int e    = idx & 15;
    int l    = (idx >> 4) & 31;
    int grp  = idx >> 9;                      // ct*K32 + kk
    int kk   = grp % K32;
    int ct   = grp / K32;
    int k    = kk * 32 + ((l >> 4) << 4) + e;
    int c    = (ct << 4) + (l & 15);
    Wf[idx] = (_Float16)W[(size_t)k * HIDC + c];
}

// ---------------------------------------------------------------------------
// GEMM: C[nrows,128] = A[nrows,K] * W[K,128]   (f16 WMMA, f32 accumulate)
// Persistent blocks: B fragments live in registers for the block's lifetime;
// A tiles are software-pipelined (global->regs while WMMA eats LDS), and the
// A fragment stream is double-buffered so ds_load overlaps the XDL pipe.
// ---------------------------------------------------------------------------
template<int K32, typename T>
__global__ __launch_bounds__(256)
void gat_gemm_wmma(const T* __restrict__ A, const _Float16* __restrict__ Wf,
                   float* __restrict__ C, int nrows)
{
    constexpr int K   = K32 * 32;
    constexpr int STR = K + 8;                // pad: 16B-aligned rows, no bank conflicts
    __shared__ _Float16 sA[16 * STR];

    const int lane = threadIdx.x & 31;
    const int wave = threadIdx.x >> 5;        // 8 waves -> 8 column tiles
    const int colbase = wave * 16;
    const int m   = lane & 15;
    const int kh  = lane >> 4;
    const int col = lane & 15;
    const int nTiles = (nrows + 15) >> 4;

    // persistent B fragments (K32 * 8 VGPRs)
    v16h b[K32];
    {
        const _Float16* wp = Wf + ((size_t)(wave * K32) * 32 + lane) * 16;
        #pragma unroll
        for (int kk = 0; kk < K32; ++kk)
            b[kk] = *(const v16h*)(wp + (size_t)kk * 32 * 16);
    }

    // A staging: thread (r = tid>>4, t = tid&15) owns row r, columns t+16j
    const int r = threadIdx.x >> 4;
    const int t = threadIdx.x & 15;
    T stage[K32 * 2];

    auto loadTile = [&](int tile) {
        int grow = (tile << 4) + r;
        if (grow < nrows) {
            const T* src = A + (size_t)grow * K + t;
            #pragma unroll
            for (int j = 0; j < K32 * 2; ++j) stage[j] = src[(size_t)j * 16];
        } else {
            #pragma unroll
            for (int j = 0; j < K32 * 2; ++j) stage[j] = (T)0.f;
        }
    };

    // A fragment from LDS (contiguous k = 16*kh + e, matching Wf): two 16B LDS loads
    auto ldA = [&](int kk) {
        const _Float16* ap = &sA[m * STR + kk * 32 + 16 * kh];
        v8h lo = *(const v8h*)ap;
        v8h hi = *(const v8h*)(ap + 8);
        v16h a;
        #pragma unroll
        for (int j = 0; j < 8; ++j) { a[j] = lo[j]; a[8 + j] = hi[j]; }
        return a;
    };

    int tile = blockIdx.x;
    if (tile >= nTiles) return;
    loadTile(tile);

    for (; tile < nTiles; tile += GEMM_GRID) {
        // commit staged rows to LDS (converted to f16)
        #pragma unroll
        for (int j = 0; j < K32 * 2; ++j)
            sA[r * STR + t + j * 16] = (_Float16)stage[j];
        __syncthreads();

        int nextTile = tile + GEMM_GRID;
        if (nextTile < nTiles) loadTile(nextTile);   // overlap with WMMA below

        int pfTile = tile + 2 * GEMM_GRID;           // prefetch one more stage ahead
        if (pfTile < nTiles)
            __builtin_prefetch(A + ((size_t)(pfTile << 4) + r) * K + t, 0, 1);

        // double-buffered A fragments: ds_load of kk+1 overlaps wmma of kk
        v8f acc = {};
        v16h acur = ldA(0);
        #pragma unroll
        for (int kk = 0; kk < K32; ++kk) {
            v16h anext = acur;
            if (kk + 1 < K32) anext = ldA(kk + 1);
            acc = __builtin_amdgcn_wmma_f32_16x16x32_f16(false, acur, false, b[kk],
                                                         (short)0, acc, false, false);
            acur = anext;
        }

        // store C: uniform full-tile fast path (always taken when nrows%16==0)
        int rowbase = tile << 4;
        float* cp = C + (size_t)(rowbase + 8 * kh) * HIDC + colbase + col;
        if (rowbase + 16 <= nrows) {
            #pragma unroll
            for (int rr = 0; rr < 8; ++rr) cp[(size_t)rr * HIDC] = acc[rr];
        } else {
            #pragma unroll
            for (int rr = 0; rr < 8; ++rr)
                if (rowbase + 8 * kh + rr < nrows) cp[(size_t)rr * HIDC] = acc[rr];
        }
        __syncthreads();
    }
}

// ---------------------------------------------------------------------------
// per-node attention logits: a_s[n,h] = <h[n,h,:], att_src[h,:]> (and a_d)
// ---------------------------------------------------------------------------
__global__ __launch_bounds__(256)
void gat_node_attn(const float* __restrict__ h,
                   const float* __restrict__ att_src, const float* __restrict__ att_dst,
                   float* __restrict__ a_s, float* __restrict__ a_d)
{
    int idx = blockIdx.x * 256 + threadIdx.x;
    if (idx >= N_NODES * NHEAD) return;
    int n = idx >> 2, hh = idx & 3;
    const float* hp = h + (size_t)n * HIDC + hh * CHEAD;
    const float* wsv = att_src + hh * CHEAD;
    const float* wdv = att_dst + hh * CHEAD;
    float s1 = 0.f, s2 = 0.f;
    #pragma unroll 8
    for (int c = 0; c < CHEAD; ++c) { float v = hp[c]; s1 += v * wsv[c]; s2 += v * wdv[c]; }
    a_s[idx] = s1;
    a_d[idx] = s2;
}

// ---------------------------------------------------------------------------
__global__ __launch_bounds__(256)
void gat_init(float* __restrict__ agg, float* __restrict__ den, unsigned* __restrict__ dmax)
{
    int i = blockIdx.x * 256 + threadIdx.x;
    if (i < N_NODES * HIDC) agg[i] = 0.f;
    if (i < N_NODES * NHEAD) { den[i] = 0.f; dmax[i] = 0u; }   // 0 < f2ord(-inf)
}

// order-preserving float<->uint encodings for atomic max
__device__ __forceinline__ unsigned f2ord(float f) {
    unsigned u = __float_as_uint(f);
    return (u & 0x80000000u) ? ~u : (u | 0x80000000u);
}
__device__ __forceinline__ float ord2f(unsigned u) {
    return (u & 0x80000000u) ? __uint_as_float(u & 0x7fffffffu) : __uint_as_float(~u);
}
__device__ __forceinline__ float leaky02(float x) { return x > 0.f ? x : 0.2f * x; }

// ---------------------------------------------------------------------------
// edge pass 1: e = leaky_relu(a_s[src]+a_d[dst]); segment max into dmax
// ---------------------------------------------------------------------------
__global__ __launch_bounds__(256)
void gat_edge_max(const float* __restrict__ a_s, const float* __restrict__ a_d,
                  const int* __restrict__ src, const int* __restrict__ dst,
                  float* __restrict__ ev, unsigned* __restrict__ dmax)
{
    int e = blockIdx.x * 256 + threadIdx.x;
    if (e >= ETOT) return;
    int s, d;
    if (e < N_EDGES) { s = src[e]; d = dst[e]; } else { s = d = e - N_EDGES; }
    float4 as = *(const float4*)(a_s + (size_t)s * 4);
    float4 ad = *(const float4*)(a_d + (size_t)d * 4);
    float4 v;
    v.x = leaky02(as.x + ad.x);
    v.y = leaky02(as.y + ad.y);
    v.z = leaky02(as.z + ad.z);
    v.w = leaky02(as.w + ad.w);
    *(float4*)(ev + (size_t)e * 4) = v;
    unsigned* mp = dmax + (size_t)d * 4;
    atomicMax(mp + 0, f2ord(v.x));
    atomicMax(mp + 1, f2ord(v.y));
    atomicMax(mp + 2, f2ord(v.z));
    atomicMax(mp + 3, f2ord(v.w));
}

// ---------------------------------------------------------------------------
// edge pass 2: ex = exp(e - m[dst]); segment sum into den; ev overwritten
// ---------------------------------------------------------------------------
__global__ __launch_bounds__(256)
void gat_edge_exp(const int* __restrict__ dst, float* __restrict__ ev,
                  const unsigned* __restrict__ dmax, float* __restrict__ den)
{
    int e = blockIdx.x * 256 + threadIdx.x;
    if (e >= ETOT) return;
    int d = (e < N_EDGES) ? dst[e] : (e - N_EDGES);
    float4 v = *(float4*)(ev + (size_t)e * 4);
    const unsigned* mp = dmax + (size_t)d * 4;
    v.x = __expf(v.x - ord2f(mp[0]));
    v.y = __expf(v.y - ord2f(mp[1]));
    v.z = __expf(v.z - ord2f(mp[2]));
    v.w = __expf(v.w - ord2f(mp[3]));
    *(float4*)(ev + (size_t)e * 4) = v;
    float* dp = den + (size_t)d * 4;
    atomicAdd(dp + 0, v.x);
    atomicAdd(dp + 1, v.y);
    atomicAdd(dp + 2, v.z);
    atomicAdd(dp + 3, v.w);
}

// ---------------------------------------------------------------------------
// edge pass 3: one wave32 per edge; lane owns 4 channels (head = lane/8):
// agg[dst] += alpha * h[src]   (float4 gather + 4 atomics per lane)
// ---------------------------------------------------------------------------
__global__ __launch_bounds__(256)
void gat_edge_aggr(const float* __restrict__ h, const float* __restrict__ ex,
                   const float* __restrict__ den,
                   const int* __restrict__ src, const int* __restrict__ dst,
                   float* __restrict__ agg)
{
    int gw = (blockIdx.x * 256 + threadIdx.x) >> 5;   // global wave = edge id
    if (gw >= ETOT) return;
    int lane = threadIdx.x & 31;
    int s, d;
    if (gw < N_EDGES) { s = src[gw]; d = dst[gw]; } else { s = d = gw - N_EDGES; }
    int c0 = lane * 4;
    int head = lane >> 3;
    float alpha = ex[(size_t)gw * 4 + head] / (den[(size_t)d * 4 + head] + 1e-16f);
    float4 hv = *(const float4*)(h + (size_t)s * HIDC + c0);
    float* out = agg + (size_t)d * HIDC + c0;
    atomicAdd(out + 0, hv.x * alpha);
    atomicAdd(out + 1, hv.y * alpha);
    atomicAdd(out + 2, hv.z * alpha);
    atomicAdd(out + 3, hv.w * alpha);
}

// ---------------------------------------------------------------------------
// epilogue: out = relu(BN(agg + bias)); OT = _Float16 (inter-layer) or float
// ---------------------------------------------------------------------------
template<typename OT>
__global__ __launch_bounds__(256)
void gat_bn_relu(const float* __restrict__ agg, const float* __restrict__ bias,
                 const float* __restrict__ gamma, const float* __restrict__ beta,
                 const float* __restrict__ mean, const float* __restrict__ var,
                 OT* __restrict__ out)
{
    int i = blockIdx.x * 256 + threadIdx.x;
    if (i >= N_NODES * HIDC) return;
    int c = i & (HIDC - 1);
    float v = agg[i] + bias[c];
    v = (v - mean[c]) * (gamma[c] * rsqrtf(var[c] + BN_EPS)) + beta[c];
    out[i] = (OT)fmaxf(v, 0.f);
}

// ---------------------------------------------------------------------------
extern "C" void kernel_launch(void* const* d_in, const int* in_sizes, int n_in,
                              void* d_out, int out_size, void* d_ws, size_t ws_size,
                              hipStream_t stream)
{
    const float* x       = (const float*)d_in[0];
    const int*   srcp    = (const int*)d_in[1];            // edge_index[0]
    const int*   dstp    = srcp + N_EDGES;                 // edge_index[1]
    const float* W1      = (const float*)d_in[2];
    const float* atts1   = (const float*)d_in[3];
    const float* attd1   = (const float*)d_in[4];
    const float* bias1   = (const float*)d_in[5];
    const float* gamma1  = (const float*)d_in[6];
    const float* beta1   = (const float*)d_in[7];
    const float* mean1   = (const float*)d_in[8];
    const float* var1    = (const float*)d_in[9];
    const float* W2      = (const float*)d_in[10];
    const float* atts2   = (const float*)d_in[11];
    const float* attd2   = (const float*)d_in[12];
    const float* bias2   = (const float*)d_in[13];
    const float* gamma2  = (const float*)d_in[14];
    const float* beta2   = (const float*)d_in[15];
    const float* mean2   = (const float*)d_in[16];
    const float* var2    = (const float*)d_in[17];

    // workspace carve-up
    float*     ws   = (float*)d_ws;
    float*     hbuf = ws;                                   // N*128 f32
    float*     agg  = hbuf + (size_t)N_NODES * HIDC;        // N*128 f32
    _Float16*  x2h  = (_Float16*)(agg + (size_t)N_NODES * HIDC);   // N*128 f16 (region sized as f32)
    float*     a_s  = (float*)x2h + (size_t)N_NODES * HIDC;        // N*4
    float*     a_d  = a_s  + (size_t)N_NODES * NHEAD;       // N*4
    float*     den  = a_d  + (size_t)N_NODES * NHEAD;       // N*4
    unsigned*  dmax = (unsigned*)(den + (size_t)N_NODES * NHEAD);  // N*4
    float*     ev   = (float*)(dmax + (size_t)N_NODES * NHEAD);    // ETOT*4
    _Float16*  wf   = (_Float16*)(ev + (size_t)ETOT * 4);   // K*128 f16 fragment-ordered

    const int gNode = (N_NODES * NHEAD + 255) / 256;
    const int gFeat = (N_NODES * HIDC + 255) / 256;
    const int gEdge = (ETOT + 255) / 256;
    const int gEdgeW = (ETOT + 7) / 8;                      // wave-per-edge, 8 waves/block

    // ---- layer 1 (K = 256, f32 input) ----
    convert_w<<<(DIN * HIDC + 255) / 256, 256, 0, stream>>>(W1, wf, DIN / 32);
    gat_gemm_wmma<DIN / 32, float><<<GEMM_GRID, 256, 0, stream>>>(x, wf, hbuf, N_NODES);
    gat_node_attn<<<gNode, 256, 0, stream>>>(hbuf, atts1, attd1, a_s, a_d);
    gat_init<<<gFeat, 256, 0, stream>>>(agg, den, dmax);
    gat_edge_max<<<gEdge, 256, 0, stream>>>(a_s, a_d, srcp, dstp, ev, dmax);
    gat_edge_exp<<<gEdge, 256, 0, stream>>>(dstp, ev, dmax, den);
    gat_edge_aggr<<<gEdgeW, 256, 0, stream>>>(hbuf, ev, den, srcp, dstp, agg);
    gat_bn_relu<_Float16><<<gFeat, 256, 0, stream>>>(agg, bias1, gamma1, beta1, mean1, var1, x2h);

    // ---- layer 2 (K = 128, f16 input) ----
    convert_w<<<(HIDC * HIDC + 255) / 256, 256, 0, stream>>>(W2, wf, HIDC / 32);
    gat_gemm_wmma<HIDC / 32, _Float16><<<GEMM_GRID, 256, 0, stream>>>(x2h, wf, hbuf, N_NODES);
    gat_node_attn<<<gNode, 256, 0, stream>>>(hbuf, atts2, attd2, a_s, a_d);
    gat_init<<<gFeat, 256, 0, stream>>>(agg, den, dmax);
    gat_edge_max<<<gEdge, 256, 0, stream>>>(a_s, a_d, srcp, dstp, ev, dmax);
    gat_edge_exp<<<gEdge, 256, 0, stream>>>(dstp, ev, dmax, den);
    gat_edge_aggr<<<gEdgeW, 256, 0, stream>>>(hbuf, ev, den, srcp, dstp, agg);
    gat_bn_relu<float><<<gFeat, 256, 0, stream>>>(agg, bias2, gamma2, beta2, mean2, var2, (float*)d_out);
}
